// LSTM1_2723009266411
// MI455X (gfx1250) — matrix operand order repaired and checked by reading, not compiled, not run on hardware
//
#include <hip/hip_runtime.h>
#include <stdint.h>

// ---------------------------------------------------------------------------
// 2-layer LSTM (B=256, T=2048, D=48, H=64) + ReLU + FC(last step) for gfx1250.
//
// MI455X reasoning: ~64 GFLOP total vs ~235 MB of HBM traffic -> far from the
// 23.3 TB/s roofline; the limiter is the serial 2x2048-step dependency chain.
// Per step a batch tile needs gates[16x256] = [x_t|h] @ [W_ih|W_hh]^T.
// Critical-path minimization:
//   * v_wmma_f32_16x16x32_bf16 (16 WMMAs/wave/step, f32 accumulate)
//   * all weight B-fragments resident in registers (loop-invariant, 128
//     VGPRs -- free at 1 wave/SIMD on a 1024-VGPR architecture)
//   * cell state c resident in accumulator-layout registers for all steps
//   * DOUBLE-BUFFERED activation tile -> exactly ONE barrier per timestep:
//     step t reads buf[t&1]; h_t and x_{t+1} are written into buf[(t+1)&1],
//     so writers never touch the buffer being read
//   * hardware V_TANH_F32 activations (sigmoid = 0.5*tanh(x/2)+0.5)
//   * global_prefetch_b8 four steps ahead for the streamed input rows
// Each wave owns gate columns {i,f,g,o} of the SAME 16 hidden units, so the
// cell update is register-local. Layer 0 writes the h1 sequence as bf16 into
// d_ws (64 MiB); layer 1 streams it back and emits the FC output.
// ---------------------------------------------------------------------------

#define H      64
#define G      256      // 4*H gates
#define T_LEN  2048
#define B_SZ   256
#define D0     48
#define AST    136      // activation row stride (ushorts); 272B: 16B-aligned, bank-spread
#define ABUF   (16 * AST)   // one activation buffer
#define WST    132      // wcat row stride (ushorts); 264B rows: bank-spread
#define CST    68       // hlast row stride (floats)

typedef __attribute__((ext_vector_type(16))) __bf16 v16bf;
typedef __attribute__((ext_vector_type(8)))  float  v8f;

union BF16Frag { v16bf v; uint32_t u[8]; };

__device__ __forceinline__ unsigned short f2bf(float f) {
  union { float f; uint32_t u; } cv; cv.f = f;
  return (unsigned short)((cv.u + 0x7FFFu + ((cv.u >> 16) & 1u)) >> 16); // RNE
}

#if __has_builtin(__builtin_amdgcn_tanhf)
__device__ __forceinline__ float tanh_f(float x) {
  return __builtin_amdgcn_tanhf(x);       // gfx1250 V_TANH_F32 transcendental
}
#else
__device__ __forceinline__ float tanh_f(float x) {
  return 1.0f - 2.0f * __builtin_amdgcn_rcpf(__expf(2.0f * x) + 1.0f);
}
#endif
__device__ __forceinline__ float sigmoid_f(float x) {
  return 0.5f * tanh_f(0.5f * x) + 0.5f;  // 1 trans op + 2 VALU
}

__global__ __launch_bounds__(128, 1)
void lstm_wmma_layer(const float* __restrict__ xin,           // layer0 in: f32 [B,T,48]
                     const unsigned short* __restrict__ hin,  // layer1 in: bf16 [B,T,64]
                     const float* __restrict__ W_ih, int Din,
                     const float* __restrict__ W_hh,
                     const float* __restrict__ b_ih,
                     const float* __restrict__ b_hh,
                     unsigned short* __restrict__ hout,       // layer0 out: bf16 [B,T,64]
                     const float* __restrict__ W_fc,
                     const float* __restrict__ b_fc,
                     float* __restrict__ out,                 // layer1 out: f32 [B]
                     int layer)
{
  // [256 gate rows][K=128]: cols 0..63 = W_ih (zero-padded past Din), 64..127 = W_hh
  __shared__ __align__(16) unsigned short wcat[G * WST];
  // double-buffered [16 rows][K=128]: cols 0..63 = x_t (padded), 64..127 = h_{t-1}
  __shared__ __align__(16) unsigned short a_lds[2 * ABUF];
  __shared__ float hlast[16 * CST];

  const int tid  = threadIdx.x;
  const int lane = tid & 31;
  const int wave = tid >> 5;            // 0..3
  const int b0   = blockIdx.x * 16;     // batch tile

  // ---- one-time: stage concatenated weights (f32 -> bf16) into LDS ----
  for (int n = tid; n < G; n += 128) {
    for (int k = 0; k < 64; ++k) {
      float w = (k < Din) ? W_ih[n * Din + k] : 0.0f;
      wcat[n * WST + k] = f2bf(w);
    }
    for (int k = 0; k < H; ++k)
      wcat[n * WST + 64 + k] = f2bf(W_hh[n * H + k]);
  }
  // zero both activation buffers (x padding + h0 = 0)
  for (int i = tid; i < 2 * ABUF; i += 128) a_lds[i] = 0;

  // per-lane fixed WMMA coordinates
  const int mA   = lane & 15;           // A-fragment row (batch row)
  const int kbA  = (lane >> 4) * 16;    // A-fragment K sub-base
  const int nC   = lane & 15;           // column within a 16-wide N tile
  const int kbB  = (lane >> 4) * 16;    // B-fragment K sub-base
  const int hcol = wave * 16 + nC;      // hidden unit 0..63 owned by this lane

  // per-lane gate biases (b_ih + b_hh), PyTorch gate order i,f,g,o
  const float bi  = b_ih[hcol]       + b_hh[hcol];
  const float bff = b_ih[64 + hcol]  + b_hh[64 + hcol];
  const float bg  = b_ih[128 + hcol] + b_hh[128 + hcol];
  const float bo  = b_ih[192 + hcol] + b_hh[192 + hcol];

  v8f c_reg = {};   // cell state for rows m = j + 8*(lane>>4), column hcol

  __syncthreads();

  // ---- hoist loop-invariant weight B-fragments into registers ----
  // 4 gates x 4 K-chunks x 8 dwords = 128 VGPRs of resident weights.
  BF16Frag wI[4], wF[4], wG[4], wO[4];
  #pragma unroll
  for (int kc = 0; kc < 4; ++kc) {
    const int koff = kc * 32;
    const int rI = (0   + hcol) * WST + koff + kbB;
    const int rF = (64  + hcol) * WST + koff + kbB;
    const int rG = (128 + hcol) * WST + koff + kbB;
    const int rO = (192 + hcol) * WST + koff + kbB;
    #pragma unroll
    for (int j = 0; j < 8; ++j) {
      wI[kc].u[j] = *(const uint32_t*)&wcat[rI + 2 * j];
      wF[kc].u[j] = *(const uint32_t*)&wcat[rF + 2 * j];
      wG[kc].u[j] = *(const uint32_t*)&wcat[rG + 2 * j];
      wO[kc].u[j] = *(const uint32_t*)&wcat[rO + 2 * j];
    }
  }

  for (int t = 0; t < T_LEN; ++t) {
    unsigned short* cur = &a_lds[(t & 1) * ABUF];        // read this step
    unsigned short* nxt = &a_lds[((t + 1) & 1) * ABUF];  // written for next step

    // ---- stage x_t tile into cur[:, 0:Din] ----
    if (layer == 0) {
      for (int idx = tid; idx < 16 * D0; idx += 128) {
        int m = idx / D0, k = idx - m * D0;
        cur[m * AST + k] =
            f2bf(xin[((size_t)(b0 + m) * T_LEN + t) * D0 + k]);
      }
      if (tid < 16 && t + 4 < T_LEN)   // global_prefetch_b8, 4 steps ahead
        __builtin_prefetch(&xin[((size_t)(b0 + tid) * T_LEN + (t + 4)) * D0], 0, 1);
    } else {
      int m = tid >> 3, hb = (tid & 7) * 8;
      uint4 v = *(const uint4*)&hin[((size_t)(b0 + m) * T_LEN + t) * H + hb];
      uint2* dst = (uint2*)&cur[m * AST + hb];
      dst[0] = make_uint2(v.x, v.y);
      dst[1] = make_uint2(v.z, v.w);
      if (tid < 16 && t + 4 < T_LEN)
        __builtin_prefetch(&hin[((size_t)(b0 + tid) * T_LEN + (t + 4)) * H], 0, 1);
    }
    // Single barrier per step: orders x_t staging (cur) and the previous
    // step's h writes (also cur, written as "nxt" last step) before reads.
    __syncthreads();

    // ---- gates = [x|h] @ [W_ih|W_hh]^T.  Wave w computes N tiles
    //      {w, w+4, w+8, w+12} = gates i/f/g/o for hidden cols [16w,16w+16).
    //      Only A-fragments touch LDS; weights come from registers.
    v8f acc_i = {}, acc_f = {}, acc_g = {}, acc_o = {};
    #pragma unroll
    for (int kc = 0; kc < 4; ++kc) {
      const int koff = kc * 32;
      BF16Frag a;
      #pragma unroll
      for (int j = 0; j < 4; ++j)
        a.u[j]     = *(const uint32_t*)&cur[mA * AST + koff + kbA + 2 * j];
      #pragma unroll
      for (int j = 0; j < 4; ++j)
        a.u[4 + j] = *(const uint32_t*)&cur[mA * AST + koff + kbA + 8 + 2 * j];

      acc_i = __builtin_amdgcn_wmma_f32_16x16x32_bf16(false, a.v, false, wI[kc].v,
                                                      (short)0, acc_i, false, false);
      acc_f = __builtin_amdgcn_wmma_f32_16x16x32_bf16(false, a.v, false, wF[kc].v,
                                                      (short)0, acc_f, false, false);
      acc_g = __builtin_amdgcn_wmma_f32_16x16x32_bf16(false, a.v, false, wG[kc].v,
                                                      (short)0, acc_g, false, false);
      acc_o = __builtin_amdgcn_wmma_f32_16x16x32_bf16(false, a.v, false, wO[kc].v,
                                                      (short)0, acc_o, false, false);
    }

    // ---- pointwise LSTM cell update: c in registers, h -> nxt buffer.
    //      No barrier needed: nxt is not read by anyone this step.
    const int mBase = (lane >> 4) * 8;
    #pragma unroll
    for (int j = 0; j < 8; ++j) {
      const int m = mBase + j;
      float iv = sigmoid_f(acc_i[j] + bi);
      float fv = sigmoid_f(acc_f[j] + bff);
      float gv = tanh_f  (acc_g[j] + bg);
      float ov = sigmoid_f(acc_o[j] + bo);
      float c  = fv * c_reg[j] + iv * gv;
      c_reg[j] = c;
      float hv = ov * tanh_f(c);
      unsigned short hb16 = f2bf(hv);
      nxt[m * AST + 64 + hcol] = hb16;            // next step's A operand
      if (layer == 0)
        hout[((size_t)(b0 + m) * T_LEN + t) * H + hcol] = hb16;  // bf16 h1 seq
      else if (t == T_LEN - 1)
        hlast[m * CST + hcol] = fmaxf(hv, 0.0f);  // relu(h_T) in f32
    }
  }

  // ---- final FC: out[b] = relu(h2_T[b]) . W_fc + b_fc ----
  if (layer == 1) {
    __syncthreads();
    if (tid < 16) {
      float acc = b_fc[0];
      #pragma unroll 8
      for (int h = 0; h < H; ++h)
        acc += hlast[tid * CST + h] * W_fc[h];
      out[b0 + tid] = acc;
    }
  }
}

extern "C" void kernel_launch(void* const* d_in, const int* in_sizes, int n_in,
                              void* d_out, int out_size, void* d_ws, size_t ws_size,
                              hipStream_t stream) {
  (void)in_sizes; (void)n_in; (void)out_size; (void)ws_size;
  const float* input = (const float*)d_in[0];
  const float* W_ih0 = (const float*)d_in[1];
  const float* W_hh0 = (const float*)d_in[2];
  const float* b_ih0 = (const float*)d_in[3];
  const float* b_hh0 = (const float*)d_in[4];
  const float* W_ih1 = (const float*)d_in[5];
  const float* W_hh1 = (const float*)d_in[6];
  const float* b_ih1 = (const float*)d_in[7];
  const float* b_hh1 = (const float*)d_in[8];
  const float* W_fc  = (const float*)d_in[9];
  const float* b_fc  = (const float*)d_in[10];

  // d_ws: bf16 h1 sequence [B,T,H] = 256*2048*64*2 = 64 MiB
  unsigned short* h1 = (unsigned short*)d_ws;
  float* out = (float*)d_out;

  dim3 grid(B_SZ / 16), block(128);
  // layer 0: f32 input -> bf16 h1 sequence in workspace
  lstm_wmma_layer<<<grid, block, 0, stream>>>(
      input, nullptr, W_ih0, D0, W_hh0, b_ih0, b_hh0,
      h1, nullptr, nullptr, nullptr, 0);
  // layer 1: bf16 h1 sequence -> final FC output
  lstm_wmma_layer<<<grid, block, 0, stream>>>(
      nullptr, h1, W_ih1, H, W_hh1, b_ih1, b_hh1,
      nullptr, W_fc, b_fc, out, 1);
}
